// TransformerLayer_11450382812010
// MI455X (gfx1250) — matrix operand order repaired and checked by reading, not compiled, not run on hardware
//
#include <hip/hip_runtime.h>
#include <hip/hip_bf16.h>

// ---------------- types ----------------
typedef __bf16 bf16;
typedef __attribute__((ext_vector_type(8)))  bf16  v8bf;
typedef __attribute__((ext_vector_type(16))) bf16  v16bf;
typedef __attribute__((ext_vector_type(8)))  float v8f;

#define EMB 1024
#define FFD 4096
#define BSZ 4
#define SEQ 2048
#define ROWS (BSZ * SEQ)   // 8192

static __device__ __forceinline__ bf16 f2b(float f) {
    unsigned u = __builtin_bit_cast(unsigned, f);
    unsigned r = u + 0x7FFFu + ((u >> 16) & 1u);   // round-to-nearest-even
    unsigned short h = (unsigned short)(r >> 16);
    return __builtin_bit_cast(bf16, h);
}
static __device__ __forceinline__ float b2f(bf16 b) {
    unsigned short h = __builtin_bit_cast(unsigned short, b);
    unsigned u = ((unsigned)h) << 16;
    return __builtin_bit_cast(float, u);
}
static __device__ __forceinline__ v16bf cat16(v8bf lo, v8bf hi) {
    return __builtin_shufflevector(lo, hi, 0,1,2,3,4,5,6,7,8,9,10,11,12,13,14,15);
}
static __device__ __forceinline__ v8f wmma_bf16(v16bf a, v16bf b, v8f c) {
    return __builtin_amdgcn_wmma_f32_16x16x32_bf16(false, a, false, b, (short)0, c, false, false);
}
static __device__ __forceinline__ v8f vzero8() {
    v8f z = {0.f,0.f,0.f,0.f,0.f,0.f,0.f,0.f};
    return z;
}

// ---------------- kernel 1: x + sinusoidal PE (f32 + bf16 copies) ----------------
__global__ __launch_bounds__(256) void pe_add_kernel(const float* __restrict__ x,
                                                     float* __restrict__ xf,
                                                     bf16* __restrict__ xb) {
    size_t i = (size_t)blockIdx.x * 256 + threadIdx.x;   // B*S*E elements
    int e = (int)(i & (EMB - 1));
    int s = (int)((i >> 10) & (SEQ - 1));
    float div = __expf(-(float)(e & ~1) * (9.210340372f / (float)EMB)); // ln(1e4)/EMB
    float ang = (float)s * div;
    float pe = (e & 1) ? __cosf(ang) : __sinf(ang);
    float v = x[i] + pe;
    xf[i] = v;
    xb[i] = f2b(v);
}

// ---------------- kernel 2: f32 -> bf16 convert ----------------
__global__ __launch_bounds__(256) void cvt_kernel(const float* __restrict__ src,
                                                  bf16* __restrict__ dst, int n) {
    int i = blockIdx.x * 256 + threadIdx.x;
    if (i < n) dst[i] = f2b(src[i]);
}

// ---------------- kernel 3: bf16 GEMM  C[M,N] = A[M,K] @ B[K,N] + bias ----------------
// block tile 128x128, 8 waves (2x4) of 64x32, K-step 32, WMMA 16x16x32 bf16.
__global__ __launch_bounds__(256) void gemm_bf16_kernel(
    const bf16* __restrict__ A, const bf16* __restrict__ B,
    const float* __restrict__ bias,
    float* __restrict__ outF, bf16* __restrict__ outB,
    int M, int N, int K, int relu)
{
    __shared__ bf16 Alds[128][32];   // [m][k]
    __shared__ bf16 Blds[128][32];   // [n][k]  (transposed stage)

    const int tid  = threadIdx.x;
    const int bm   = blockIdx.y * 128;
    const int bn   = blockIdx.x * 128;
    const int wid  = tid >> 5;
    const int lane = tid & 31;
    const int wm   = (wid >> 2) * 64;   // wave row offset
    const int wn   = (wid & 3) * 32;    // wave col offset
    const int col  = lane & 15;
    const int hlf  = lane >> 4;

    v8f acc[4][2];
#pragma unroll
    for (int m = 0; m < 4; ++m)
#pragma unroll
        for (int n = 0; n < 2; ++n) acc[m][n] = vzero8();

    for (int k0 = 0; k0 < K; k0 += 32) {
        // stage A tile (128x32): 512 chunks of 8 bf16, 2 per thread
#pragma unroll
        for (int i = 0; i < 2; ++i) {
            int c  = tid * 2 + i;
            int r  = c >> 2;
            int kk = (c & 3) * 8;
            v8bf v = *(const v8bf*)(A + (size_t)(bm + r) * K + k0 + kk);
            *(v8bf*)(&Alds[r][kk]) = v;
            if (k0 + 32 < K)
                __builtin_prefetch(A + (size_t)(bm + r) * K + k0 + 32 + kk, 0, 1);
        }
        // stage B tile (32x128 -> Blds[n][k]): 512 chunks of 8 along n
#pragma unroll
        for (int i = 0; i < 2; ++i) {
            int c  = tid * 2 + i;
            int kk = c >> 4;
            int n0 = (c & 15) * 8;
            v8bf v = *(const v8bf*)(B + (size_t)(k0 + kk) * N + bn + n0);
#pragma unroll
            for (int j = 0; j < 8; ++j) Blds[n0 + j][kk] = v[j];
            if (k0 + 32 < K)
                __builtin_prefetch(B + (size_t)(k0 + 32 + kk) * N + bn + n0, 0, 1);
        }
        __syncthreads();

        // B fragments: lane = col n, halves hold K 0..15 / 16..31
        v16bf bfrag[2];
#pragma unroll
        for (int nt = 0; nt < 2; ++nt) {
            const bf16* bp = &Blds[wn + nt * 16 + col][hlf * 16];
            bfrag[nt] = cat16(*(const v8bf*)bp, *(const v8bf*)(bp + 8));
        }
        // A fragments: lane = row m, halves hold K 0..15 / 16..31
#pragma unroll
        for (int mt = 0; mt < 4; ++mt) {
            const bf16* ap = &Alds[wm + mt * 16 + col][hlf * 16];
            v16bf afrag = cat16(*(const v8bf*)ap, *(const v8bf*)(ap + 8));
            acc[mt][0] = wmma_bf16(afrag, bfrag[0], acc[mt][0]);
            acc[mt][1] = wmma_bf16(afrag, bfrag[1], acc[mt][1]);
        }
        __syncthreads();
    }

    // epilogue: C layout -> row = vr + 8*hlf, col = lane&15
#pragma unroll
    for (int mt = 0; mt < 4; ++mt)
#pragma unroll
        for (int nt = 0; nt < 2; ++nt)
#pragma unroll
            for (int vr = 0; vr < 8; ++vr) {
                int row = bm + wm + mt * 16 + vr + hlf * 8;
                int cn  = bn + wn + nt * 16 + col;
                float v = acc[mt][nt][vr] + bias[cn];
                if (relu) v = fmaxf(v, 0.f);
                size_t o = (size_t)row * N + cn;
                if (outF) outF[o] = v;
                if (outB) outB[o] = f2b(v);
            }
}

// ---------------- kernel 4: fused attention ----------------
// One block (8 waves) per 16-query tile. Scores -> LDS, block softmax, P@V.
__global__ __launch_bounds__(256) void attn_kernel(
    const bf16* __restrict__ Qg, const bf16* __restrict__ Kg,
    const bf16* __restrict__ Vg, float* __restrict__ O)
{
    extern __shared__ char smem[];
    bf16*  Qs   = (bf16*)smem;                    // [16][1024]  32KB
    bf16*  Ps   = (bf16*)(smem + 32768);          // [16][2048]  64KB
    bf16*  Vt   = (bf16*)(smem + 98304);          // [1024][32]  64KB  (n-major)
    float* linv = (float*)(smem + 163840);        // [16]

    const int tid   = threadIdx.x;
    const int wid   = tid >> 5;
    const int lane  = tid & 31;
    const int col   = lane & 15;
    const int hlf   = lane >> 4;
    const int batch = blockIdx.x / (SEQ / 16);
    const int q0    = (blockIdx.x % (SEQ / 16)) * 16;

    const bf16* Qb = Qg + (size_t)batch * SEQ * EMB;
    const bf16* Kb = Kg + (size_t)batch * SEQ * EMB;
    const bf16* Vb = Vg + (size_t)batch * SEQ * EMB;

    // ---- stage Q tile 16x1024 ----
#pragma unroll
    for (int i = 0; i < 8; ++i) {
        int c  = tid + i * 256;          // 2048 chunks of 8
        int r  = c >> 7;
        int e0 = (c & 127) * 8;
        *(v8bf*)(Qs + r * EMB + e0) = *(const v8bf*)(Qb + (size_t)(q0 + r) * EMB + e0);
    }
    __syncthreads();

    // ---- phase 1: scores S = Q @ K^T / 32, park bf16 in Ps ----
    for (int kt = wid; kt < SEQ / 16; kt += 8) {
        int k0g = kt * 16;
        v8f sc = vzero8();
#pragma unroll 4
        for (int ec = 0; ec < EMB / 32; ++ec) {
            int e0 = ec * 32;
            const bf16* kp = Kb + (size_t)(k0g + col) * EMB + e0 + hlf * 16;
            v16bf bfrag = cat16(*(const v8bf*)kp, *(const v8bf*)(kp + 8));
            const bf16* qp = Qs + col * EMB + e0 + hlf * 16;
            v16bf afrag = cat16(*(const v8bf*)qp, *(const v8bf*)(qp + 8));
            sc = wmma_bf16(afrag, bfrag, sc);
        }
#pragma unroll
        for (int vr = 0; vr < 8; ++vr)
            Ps[(vr + hlf * 8) * SEQ + k0g + col] = f2b(sc[vr] * 0.03125f);
    }
    __syncthreads();

    // ---- phase 2: softmax over 2048 keys; wave w handles rows 2w, 2w+1 ----
    {
        int row = 2 * wid + hlf;
        float m = -3.0e38f;
        for (int t = 0; t < SEQ / 16; ++t)
            m = fmaxf(m, b2f(Ps[row * SEQ + col + t * 16]));
#pragma unroll
        for (int off = 1; off < 16; off <<= 1) m = fmaxf(m, __shfl_xor(m, off, 16));
        float sum = 0.f;
        for (int t = 0; t < SEQ / 16; ++t) {
            int idx = row * SEQ + col + t * 16;
            float p = __expf(b2f(Ps[idx]) - m);
            sum += p;
            Ps[idx] = f2b(p);
        }
#pragma unroll
        for (int off = 1; off < 16; off <<= 1) sum += __shfl_xor(sum, off, 16);
        if (col == 0) linv[row] = 1.f / sum;
    }
    __syncthreads();

    // ---- phase 3: O = (P @ V) * linv; wave w covers cols w*128..w*128+127 ----
    v8f o[8];
#pragma unroll
    for (int nt = 0; nt < 8; ++nt) o[nt] = vzero8();

    for (int kc = 0; kc < SEQ / 32; ++kc) {
        __syncthreads();
        // stage V chunk [32 keys x 1024] transposed -> Vt[n][kk]
#pragma unroll
        for (int i = 0; i < 16; ++i) {
            int c  = tid + i * 256;       // 4096 chunks of 8
            int kk = c >> 7;
            int n0 = (c & 127) * 8;
            v8bf v = *(const v8bf*)(Vb + (size_t)(kc * 32 + kk) * EMB + n0);
#pragma unroll
            for (int j = 0; j < 8; ++j) Vt[(n0 + j) * 32 + kk] = v[j];
        }
        __syncthreads();

        const bf16* pp = Ps + col * SEQ + kc * 32 + hlf * 16;
        v16bf afrag = cat16(*(const v8bf*)pp, *(const v8bf*)(pp + 8));
#pragma unroll
        for (int nt = 0; nt < 8; ++nt) {
            const bf16* vp = Vt + (wid * 128 + nt * 16 + col) * 32 + hlf * 16;
            v16bf bfrag = cat16(*(const v8bf*)vp, *(const v8bf*)(vp + 8));
            o[nt] = wmma_bf16(afrag, bfrag, o[nt]);
        }
    }

#pragma unroll
    for (int nt = 0; nt < 8; ++nt)
#pragma unroll
        for (int vr = 0; vr < 8; ++vr) {
            int rl   = vr + hlf * 8;
            int gcol = wid * 128 + nt * 16 + col;
            O[(size_t)batch * SEQ * EMB + (size_t)(q0 + rl) * EMB + gcol] =
                o[nt][vr] * linv[rl];
        }
}

// ---------------- kernel 5: y = LayerNorm(acc + res) * g + be ----------------
__global__ __launch_bounds__(256) void add_ln_kernel(
    const float* __restrict__ acc, const float* __restrict__ res,
    const float* __restrict__ g, const float* __restrict__ be,
    float* __restrict__ outF, bf16* __restrict__ outB)
{
    __shared__ float red[8];
    const int row  = blockIdx.x;
    const int wid  = threadIdx.x >> 5;
    const int lane = threadIdx.x & 31;
    const float* a = acc + (size_t)row * EMB;
    const float* r = res + (size_t)row * EMB;

    float v[4];
    float s = 0.f;
#pragma unroll
    for (int i = 0; i < 4; ++i) {
        int e = threadIdx.x + i * 256;
        v[i] = a[e] + r[e];
        s += v[i];
    }
#pragma unroll
    for (int off = 16; off > 0; off >>= 1) s += __shfl_xor(s, off, 32);
    if (lane == 0) red[wid] = s;
    __syncthreads();
    float tot = 0.f;
#pragma unroll
    for (int i = 0; i < 8; ++i) tot += red[i];
    float mu = tot * (1.f / EMB);
    __syncthreads();

    float s2 = 0.f;
#pragma unroll
    for (int i = 0; i < 4; ++i) { float d = v[i] - mu; s2 += d * d; }
#pragma unroll
    for (int off = 16; off > 0; off >>= 1) s2 += __shfl_xor(s2, off, 32);
    if (lane == 0) red[wid] = s2;
    __syncthreads();
    float tv = 0.f;
#pragma unroll
    for (int i = 0; i < 8; ++i) tv += red[i];
    float rs = rsqrtf(tv * (1.f / EMB) + 1e-5f);

#pragma unroll
    for (int i = 0; i < 4; ++i) {
        int e = threadIdx.x + i * 256;
        float o = (v[i] - mu) * rs * g[e] + be[e];
        size_t idx = (size_t)row * EMB + e;
        outF[idx] = o;
        if (outB) outB[idx] = f2b(o);
    }
}

// ---------------- host launcher ----------------
extern "C" void kernel_launch(void* const* d_in, const int* in_sizes, int n_in,
                              void* d_out, int out_size, void* d_ws, size_t ws_size,
                              hipStream_t stream) {
    (void)in_sizes; (void)n_in; (void)out_size; (void)ws_size;

    const float* x   = (const float*)d_in[0];
    const float* Wq  = (const float*)d_in[1];
    const float* bq  = (const float*)d_in[2];
    const float* Wk  = (const float*)d_in[3];
    const float* bk  = (const float*)d_in[4];
    const float* Wv  = (const float*)d_in[5];
    const float* bv  = (const float*)d_in[6];
    const float* W1  = (const float*)d_in[7];
    const float* b1  = (const float*)d_in[8];
    const float* W2  = (const float*)d_in[9];
    const float* b2  = (const float*)d_in[10];
    const float* g1  = (const float*)d_in[11];
    const float* be1 = (const float*)d_in[12];
    const float* g2  = (const float*)d_in[13];
    const float* be2 = (const float*)d_in[14];
    float* out = (float*)d_out;

    char* ws = (char*)d_ws;
    size_t off = 0;
    auto take = [&](size_t bytes) { char* p = ws + off; off += (bytes + 255) & ~(size_t)255; return p; };

    float* xpe_f  = (float*)take((size_t)ROWS * EMB * 4);
    bf16*  xpe_b  = (bf16*) take((size_t)ROWS * EMB * 2);
    bf16*  Wq_b   = (bf16*) take((size_t)EMB * EMB * 2);
    bf16*  Wk_b   = (bf16*) take((size_t)EMB * EMB * 2);
    bf16*  Wv_b   = (bf16*) take((size_t)EMB * EMB * 2);
    bf16*  W1_b   = (bf16*) take((size_t)EMB * FFD * 2);
    bf16*  W2_b   = (bf16*) take((size_t)FFD * EMB * 2);
    bf16*  Q_b    = (bf16*) take((size_t)ROWS * EMB * 2);
    bf16*  K_b    = (bf16*) take((size_t)ROWS * EMB * 2);
    bf16*  V_b    = (bf16*) take((size_t)ROWS * EMB * 2);
    float* h_f    = (float*)take((size_t)ROWS * EMB * 4);
    bf16*  h_b    = (bf16*) take((size_t)ROWS * EMB * 2);
    bf16*  ff1_b  = (bf16*) take((size_t)ROWS * FFD * 2);

    // 1) positional encoding + bf16 copy
    pe_add_kernel<<<(ROWS * EMB) / 256, 256, 0, stream>>>(x, xpe_f, xpe_b);

    // 2) weight conversion
    cvt_kernel<<<(EMB * EMB) / 256, 256, 0, stream>>>(Wq, Wq_b, EMB * EMB);
    cvt_kernel<<<(EMB * EMB) / 256, 256, 0, stream>>>(Wk, Wk_b, EMB * EMB);
    cvt_kernel<<<(EMB * EMB) / 256, 256, 0, stream>>>(Wv, Wv_b, EMB * EMB);
    cvt_kernel<<<(EMB * FFD) / 256, 256, 0, stream>>>(W1, W1_b, EMB * FFD);
    cvt_kernel<<<(FFD * EMB) / 256, 256, 0, stream>>>(W2, W2_b, FFD * EMB);

    // 3) Q/K/V projections (bf16 outputs)
    gemm_bf16_kernel<<<dim3(EMB / 128, ROWS / 128), 256, 0, stream>>>(
        xpe_b, Wq_b, bq, nullptr, Q_b, ROWS, EMB, EMB, 0);
    gemm_bf16_kernel<<<dim3(EMB / 128, ROWS / 128), 256, 0, stream>>>(
        xpe_b, Wk_b, bk, nullptr, K_b, ROWS, EMB, EMB, 0);
    gemm_bf16_kernel<<<dim3(EMB / 128, ROWS / 128), 256, 0, stream>>>(
        xpe_b, Wv_b, bv, nullptr, V_b, ROWS, EMB, EMB, 0);

    // 4) fused attention -> h_f
    attn_kernel<<<ROWS / 16, 256, 163904, stream>>>(Q_b, K_b, V_b, h_f);

    // 5) h = LN(attn + xpe)
    add_ln_kernel<<<ROWS, 256, 0, stream>>>(h_f, xpe_f, g1, be1, h_f, h_b);

    // 6) FF1 (ReLU, bf16 out)
    gemm_bf16_kernel<<<dim3(FFD / 128, ROWS / 128), 256, 0, stream>>>(
        h_b, W1_b, b1, nullptr, ff1_b, ROWS, FFD, EMB, 1);

    // 7) FF2 (f32 out -> d_out)
    gemm_bf16_kernel<<<dim3(EMB / 128, ROWS / 128), 256, 0, stream>>>(
        ff1_b, W2_b, b2, out, nullptr, ROWS, EMB, FFD, 0);

    // 8) out = LN(ff2 + h)
    add_ln_kernel<<<ROWS, 256, 0, stream>>>(out, h_f, g2, be2, out, nullptr);
}
